// ST_LSTMCell_21534966022556
// MI455X (gfx1250) — compile-verified, matrix operand order checked by reading, not run on hardware
//
#include <hip/hip_runtime.h>
#include <hip/hip_bf16.h>

// ---------------------------------------------------------------------------
// ST-LSTM cell, fused:  gates = [x|h_t|h_s] @ [U;Wt;Ws] + b  (BF16 WMMA, f32 acc)
// then per-element: i,fs,ft,o = sigmoid, c_n = tanh, c_h = i*c_n + ft*c_t + fs*c_s,
// h = o * tanh(c_h).  Outputs concatenated: [h (B*H) | c_h (B*H)] fp32.
//
// Data path: async global->LDS copies (ASYNCcnt) with double-buffered stages,
// falling back to global_load_b128 + ds_store_b128 if the builtin is absent.
// ---------------------------------------------------------------------------

#define BB   8192
#define DD   256
#define HH   512
#define KK   1280          // D + 2H
#define NG   5             // gates: i, fs, ft, o, c
#define KS   64            // K elements per LDS stage (2 WMMA K-steps)
#define LDT  72            // padded LDS row stride (bf16 elems): 144B -> no 8-way conflicts

typedef __attribute__((ext_vector_type(16))) __bf16 v16bf;
typedef __attribute__((ext_vector_type(8)))  __bf16 v8bf;
typedef __attribute__((ext_vector_type(8)))  float  v8f;
typedef __attribute__((ext_vector_type(4)))  int    v4i;

#if defined(__has_builtin)
#if __has_builtin(__builtin_amdgcn_global_load_async_to_lds_b128) && \
    __has_builtin(__builtin_amdgcn_s_wait_asynccnt)
#define USE_ASYNC 1
#endif
#endif

#ifdef USE_ASYNC
#define ASYNC_WAIT(n) __builtin_amdgcn_s_wait_asynccnt(n)
#else
#define ASYNC_WAIT(n)
#endif

// copy 16B (8 bf16) global -> LDS
__device__ __forceinline__ void cp16(const __bf16* g, __bf16* l) {
#ifdef USE_ASYNC
    typedef __attribute__((address_space(1))) v4i GV;   // global v4i
    typedef __attribute__((address_space(3))) v4i LV;   // LDS v4i
    __builtin_amdgcn_global_load_async_to_lds_b128((GV*)g, (LV*)l, 0, 0);
#else
    *(v8bf*)l = *(const v8bf*)g;
#endif
}

// ---------------- pack kernels (one-time fp32 -> bf16 conversion) ----------

__global__ __launch_bounds__(256) void pack_A(const float* __restrict__ x,
                                              const float* __restrict__ ht,
                                              const float* __restrict__ hs,
                                              __bf16* __restrict__ Abf) {
    size_t t   = (size_t)blockIdx.x * 256 + threadIdx.x;   // one thread per 8 elems
    size_t idx = t * 8;
    size_t b   = idx / KK;
    int    k   = (int)(idx % KK);
    const float* src;
    if (k < DD)            src = x  + b * DD + k;
    else if (k < DD + HH)  src = ht + b * HH + (k - DD);
    else                   src = hs + b * HH + (k - DD - HH);
    v8bf o;
#pragma unroll
    for (int j = 0; j < 8; ++j) o[j] = (__bf16)src[j];
    *(v8bf*)(Abf + idx) = o;
}

struct WPtrs { const float* u[NG]; const float* wt[NG]; const float* ws[NG]; };

// Writes W^T : Wp[col][k], col = g*H + n, k in [0,1280)
__global__ __launch_bounds__(256) void pack_W(WPtrs w, __bf16* __restrict__ Wp) {
    size_t t   = (size_t)blockIdx.x * 256 + threadIdx.x;   // one thread per 8 k-elems
    int    col = (int)(t / (KK / 8));
    int    k   = (int)(t % (KK / 8)) * 8;
    int    g   = col / HH;
    int    n   = col % HH;
    v8bf o;
#pragma unroll
    for (int j = 0; j < 8; ++j) {
        int kk = k + j;
        float f;
        if (kk < DD)           f = w.u [g][(size_t)kk * HH + n];
        else if (kk < DD + HH) f = w.wt[g][(size_t)(kk - DD) * HH + n];
        else                   f = w.ws[g][(size_t)(kk - DD - HH) * HH + n];
        o[j] = (__bf16)f;
    }
    *(v8bf*)(Wp + (size_t)col * KK + k) = o;
}

struct BPtrs { const float* b[NG]; };

__global__ __launch_bounds__(256) void pack_bias(BPtrs bp, float* __restrict__ bias) {
    int t = blockIdx.x * 256 + threadIdx.x;
    if (t < NG * HH) bias[t] = bp.b[t / HH][t % HH];
}

// ---------------- fused GEMM + LSTM epilogue -------------------------------

__device__ __forceinline__ float sigmoid_f(float x) {
    return 1.0f / (1.0f + __expf(-x));
}
__device__ __forceinline__ float tanh_f(float x) {
    float e = __expf(-2.0f * __builtin_fabsf(x));
    float t = (1.0f - e) / (1.0f + e);
    return __builtin_copysignf(t, x);
}

__global__ __launch_bounds__(256) void stlstm_gemm(
        const __bf16* __restrict__ Abf,   // [B][K] bf16
        const __bf16* __restrict__ Wp,    // [5H][K] bf16 (W transposed)
        const float*  __restrict__ bias,  // [5H]
        const float*  __restrict__ c_t,   // [B][H]
        const float*  __restrict__ c_s,   // [B][H]
        float*        __restrict__ out)   // [h (B*H) | c_h (B*H)]
{
    __shared__ __align__(16) __bf16 As[2][64 * LDT];        // 18 KB
    __shared__ __align__(16) __bf16 Bs[2][NG][64 * LDT];    // 90 KB

    const int tid  = threadIdx.x;
    const int lane = tid & 31;
    const int wave = tid >> 5;       // 0..7
    const int wm   = wave & 1;       // wave row (M): 0..1  -> 32 rows each
    const int wn   = wave >> 1;      // wave col (N): 0..3  -> 16 cols each

    const int m_base = blockIdx.x * 64;      // B/64 = 128 blocks
    const int n_base = blockIdx.y * 64;      // H/64 = 8 blocks

    const int l16 = lane & 15;
    const int kh  = (lane >> 4) << 3;        // 0 for lanes 0-15, 8 for lanes 16-31

    v8f acc[2][NG];
#pragma unroll
    for (int i = 0; i < 2; ++i)
#pragma unroll
        for (int g = 0; g < NG; ++g)
            acc[i][g] = (v8f){0.f, 0.f, 0.f, 0.f, 0.f, 0.f, 0.f, 0.f};

    // fill mapping: 512 chunks of 16B per tile; each thread owns chunks tid, tid+256
    // chunk c: row = c>>3 (0..63), grp = (c&7)*8 (0..56)
    // per stage each thread issues 1+NG async copies per chunk -> 12 total
    auto fill_stage = [&](int k0, int buf) {
#pragma unroll
        for (int h = 0; h < 2; ++h) {
            int c   = tid + (h << 8);
            int row = c >> 3;
            int grp = (c & 7) << 3;
            cp16(Abf + (size_t)(m_base + row) * KK + k0 + grp,
                 &As[buf][row * LDT + grp]);
#pragma unroll
            for (int g = 0; g < NG; ++g)
                cp16(Wp + (size_t)(g * HH + n_base + row) * KK + k0 + grp,
                     &Bs[buf][g][row * LDT + grp]);
        }
    };

    const int NST = KK / KS;   // 20 stages
    fill_stage(0, 0);

    for (int s = 0; s < NST; ++s) {
        const int cur = s & 1;
        if (s + 1 < NST) {
            fill_stage((s + 1) * KS, cur ^ 1);   // prefetch next stage
            ASYNC_WAIT(12);                      // stage s complete (in-order dones)
        } else {
            ASYNC_WAIT(0);
        }
        __syncthreads();                         // stage s visible to all waves

#pragma unroll
        for (int kk = 0; kk < KS; kk += 32) {
            v16bf afr[2];
#pragma unroll
            for (int i = 0; i < 2; ++i) {
                int r = wm * 32 + i * 16 + l16;
                union { v16bf v; v8bf h[2]; } u;
                u.h[0] = *(const v8bf*)&As[cur][r * LDT + kk + kh];
                u.h[1] = *(const v8bf*)&As[cur][r * LDT + kk + kh + 16];
                afr[i] = u.v;
            }
#pragma unroll
            for (int g = 0; g < NG; ++g) {
                int c = wn * 16 + l16;
                union { v16bf v; v8bf h[2]; } u;
                u.h[0] = *(const v8bf*)&Bs[cur][g][c * LDT + kk + kh];
                u.h[1] = *(const v8bf*)&Bs[cur][g][c * LDT + kk + kh + 16];
                v16bf bfr = u.v;
                acc[0][g] = __builtin_amdgcn_wmma_f32_16x16x32_bf16(
                    false, afr[0], false, bfr, (short)0, acc[0][g], false, false);
                acc[1][g] = __builtin_amdgcn_wmma_f32_16x16x32_bf16(
                    false, afr[1], false, bfr, (short)0, acc[1][g], false, false);
            }
        }
        __syncthreads();                         // all waves done reading buf `cur`
    }

    // ---------------- epilogue: activations + cell blend -------------------
    const int n_g = n_base + wn * 16 + l16;          // 0..511
    const float b_i  = bias[0 * HH + n_g];
    const float b_fs = bias[1 * HH + n_g];
    const float b_ft = bias[2 * HH + n_g];
    const float b_o  = bias[3 * HH + n_g];
    const float b_c  = bias[4 * HH + n_g];

#pragma unroll
    for (int i = 0; i < 2; ++i) {
#pragma unroll
        for (int v = 0; v < 8; ++v) {
            int m_g = m_base + wm * 32 + i * 16 + v + kh;  // C/D layout: M = v (+8 for hi lanes)
            size_t idx = (size_t)m_g * HH + n_g;

            float gi  = acc[i][0][v] + b_i;
            float gfs = acc[i][1][v] + b_fs;
            float gft = acc[i][2][v] + b_ft;
            float go  = acc[i][3][v] + b_o;
            float gc  = acc[i][4][v] + b_c;

            float in_ = sigmoid_f(gi);
            float fs  = sigmoid_f(gfs);
            float ft  = sigmoid_f(gft);
            float oo  = sigmoid_f(go);
            float cn  = tanh_f(gc);

            float ch = in_ * cn + ft * c_t[idx] + fs * c_s[idx];
            out[idx]                    = oo * tanh_f(ch);   // h
            out[(size_t)BB * HH + idx]  = ch;                // c_h
        }
    }
}

// ---------------------------------------------------------------------------

extern "C" void kernel_launch(void* const* d_in, const int* in_sizes, int n_in,
                              void* d_out, int out_size, void* d_ws, size_t ws_size,
                              hipStream_t stream) {
    (void)in_sizes; (void)n_in; (void)out_size; (void)ws_size;

    const float* x   = (const float*)d_in[0];
    const float* h_t = (const float*)d_in[1];
    const float* h_s = (const float*)d_in[2];
    const float* c_t = (const float*)d_in[3];
    const float* c_s = (const float*)d_in[4];

    WPtrs wp; BPtrs bp;
    for (int g = 0; g < NG; ++g) {
        wp.u [g] = (const float*)d_in[5 + 4 * g];
        wp.wt[g] = (const float*)d_in[6 + 4 * g];
        wp.ws[g] = (const float*)d_in[7 + 4 * g];
        bp.b [g] = (const float*)d_in[8 + 4 * g];
    }

    // workspace layout
    char* ws = (char*)d_ws;
    __bf16* Abf  = (__bf16*)(ws);                                   // 8192*1280*2  = 20.0 MB
    __bf16* Wpk  = (__bf16*)(ws + (size_t)BB * KK * 2);             // 2560*1280*2  =  6.25 MB
    float*  bias = (float*) (ws + (size_t)BB * KK * 2
                                + (size_t)NG * HH * KK * 2);        // 5*512*4

    float* out = (float*)d_out;

    // pack (one-time conversions)
    pack_A   <<<(BB * KK / 8) / 256, 256, 0, stream>>>(x, h_t, h_s, Abf);
    pack_W   <<<(NG * HH * KK / 8) / 256, 256, 0, stream>>>(wp, Wpk);
    pack_bias<<<(NG * HH + 255) / 256, 256, 0, stream>>>(bp, bias);

    // fused GEMM + LSTM epilogue
    dim3 grid(BB / 64, HH / 64);
    stlstm_gemm<<<grid, 256, 0, stream>>>(Abf, Wpk, bias, c_t, c_s, out);
}
